// TargetHead_18236431139263
// MI455X (gfx1250) — compile-verified, matrix operand order checked by reading, not run on hardware
//
#include <hip/hip_runtime.h>
#include <hip/hip_bf16.h>

// ---------------- types ----------------
typedef __attribute__((ext_vector_type(16))) __bf16          v16bf;
typedef __attribute__((ext_vector_type(8)))  float           v8f;
typedef __attribute__((ext_vector_type(4)))  float           f32x4;
typedef __attribute__((ext_vector_type(8)))  unsigned short  u16x8;
typedef __attribute__((ext_vector_type(16))) unsigned short  u16x16;

#define B_   64
#define U_   512
#define AE_  1024
#define UE_  256
#define A_   256
#define KQ_  (AE_ + UE_)   // 1280

// f32 -> bf16 via hardware convert (fptrunc lowers to v_cvt_*bf16_f32, RNE)
__device__ __forceinline__ unsigned short f2bf(float f) {
    return __builtin_bit_cast(unsigned short, (__bf16)f);
}

// Load a 16-element bf16 WMMA fragment for this lane: two contiguous runs of
// 8 bf16 (16B each) at p and p+16 elements. Matches the CDNA5 16-bit operand
// layout when the caller offsets p by (kc + kbase) with kbase = (lane<16)?0:8.
__device__ __forceinline__ v16bf load_bf16_frag(const unsigned short* p) {
    u16x8 lo = *(const u16x8*)(p);
    u16x8 hi = *(const u16x8*)(p + 16);
    u16x16 v;
#pragma unroll
    for (int e = 0; e < 8; ++e) { v[e] = lo[e]; v[8 + e] = hi[e]; }
    return __builtin_bit_cast(v16bf, v);
}

__device__ __forceinline__ v8f zero_v8f() {
    v8f z;
#pragma unroll
    for (int e = 0; e < 8; ++e) z[e] = 0.0f;
    return z;
}

// ---------------- kernel 1: weights -> bf16, transposed to [A][K] ----------------
__global__ void TargetHead_wtrans_kernel(const float* __restrict__ Wq,
                                         const float* __restrict__ Wk,
                                         unsigned short* __restrict__ WqT,
                                         unsigned short* __restrict__ WkT) {
    int i = blockIdx.x * blockDim.x + threadIdx.x;
    const int nq = A_ * KQ_;
    if (i < nq) {
        int a = i / KQ_, k = i % KQ_;
        WqT[i] = f2bf(Wq[(size_t)k * A_ + a]);
    } else {
        int j = i - nq;
        if (j < A_ * UE_) {
            int a = j / UE_, k = j % UE_;
            WkT[j] = f2bf(Wk[(size_t)k * A_ + a]);
        }
    }
}

// ---------------- kernel 2: Q = combined@Wq + bq ; K = enemy@Wk + bk (bf16 out) ----------------
__global__ void __launch_bounds__(32)
TargetHead_qk_gemm_kernel(const float* __restrict__ are,    // [B,U,AE]
                          const float* __restrict__ own,    // [B,U,UE]
                          const float* __restrict__ enemy,  // [B,U,UE]
                          const float* __restrict__ bq,
                          const float* __restrict__ bk,
                          const unsigned short* __restrict__ WqT,  // [A][KQ]
                          const unsigned short* __restrict__ WkT,  // [A][UE]
                          unsigned short* __restrict__ Qb,   // [B,U,A] bf16
                          unsigned short* __restrict__ Kb) { // [B,U,A] bf16
    const int lane  = threadIdx.x;
    const int isK   = blockIdx.y;              // 0 = Q path, 1 = K path
    const int b     = blockIdx.x / (U_ / 16);
    const int ut    = blockIdx.x % (U_ / 16);
    const int u0    = ut * 16;

    const int m     = lane & 15;
    const int kbase = (lane >> 4) << 3;        // 0 or 8 (also the D-row offset)
    const int row   = u0 + m;

    const int Kdim                 = isK ? UE_ : KQ_;
    const unsigned short* WT       = isK ? WkT : WqT;
    const float* bias              = isK ? bk  : bq;
    unsigned short* Out            = isK ? Kb  : Qb;

    v8f acc[16];
#pragma unroll
    for (int t = 0; t < 16; ++t) acc[t] = zero_v8f();

    for (int kc = 0; kc < Kdim; kc += 32) {
        const int k0 = kc + kbase;

        // prefetch next weight chunk into cache (global_prefetch_b8)
        __builtin_prefetch(WT + (size_t)m * Kdim + kc + 64, 0, 1);

        // ---- A fragment: 16 bf16 from two runs of 8 f32 (hardware cvt, RNE)
        v16bf a;
#pragma unroll
        for (int half = 0; half < 2; ++half) {
            const int ks = k0 + half * 16;
            const float* src;
            if (isK) {
                src = enemy + ((size_t)(b * U_ + row)) * UE_ + ks;
            } else if (ks < AE_) {
                src = are + ((size_t)(b * U_ + row)) * AE_ + ks;
            } else {
                src = own + ((size_t)(b * U_ + row)) * UE_ + (ks - AE_);
            }
            f32x4 p0 = *(const f32x4*)(src);
            f32x4 p1 = *(const f32x4*)(src + 4);
#pragma unroll
            for (int e = 0; e < 4; ++e) {
                a[half * 8 + e]     = (__bf16)p0[e];
                a[half * 8 + 4 + e] = (__bf16)p1[e];
            }
        }

        // ---- 16 output tiles across A=256
#pragma unroll
        for (int t = 0; t < 16; ++t) {
            const unsigned short* wrow = WT + (size_t)(t * 16 + m) * Kdim + k0;
            v16bf bf = load_bf16_frag(wrow);
            acc[t] = __builtin_amdgcn_wmma_f32_16x16x32_bf16(
                false, a, false, bf, (short)0, acc[t], false, false);
        }
    }

    // bias + convert + store (bf16, [B,U,A] row-major)
#pragma unroll
    for (int t = 0; t < 16; ++t) {
        const int colg = t * 16 + m;
        const float bv = bias[colg];
#pragma unroll
        for (int r = 0; r < 8; ++r) {
            const int rowg = u0 + r + kbase;
            Out[((size_t)(b * U_ + rowg)) * A_ + colg] = f2bf(acc[t][r] + bv);
        }
    }
}

// ---------------- kernel 3: S = Q@K^T/16, mask, softmax ----------------
__global__ void __launch_bounds__(128)
TargetHead_attn_softmax_kernel(const unsigned short* __restrict__ Qb,  // [B,U,A] bf16
                               const unsigned short* __restrict__ Kb,  // [B,U,A] bf16
                               const int* __restrict__ nr_own,
                               const int* __restrict__ nr_enemy,
                               const int* __restrict__ nr_flags,
                               float* __restrict__ out) {              // [B,U,U]
    __shared__ float sS[16 * U_];       // 32 KB logit slab
    __shared__ float sred[16 * 8];      // per-row partial reductions

    const int tid  = threadIdx.x;
    const int lane = tid & 31;
    const int w    = tid >> 5;          // wave 0..3, owns cols [w*128, w*128+128)
    const int b    = blockIdx.x / (U_ / 16);
    const int ut   = blockIdx.x % (U_ / 16);
    const int u0   = ut * 16;

    const int m     = lane & 15;
    const int kbase = (lane >> 4) << 3;

    const unsigned short* Qbase = Qb + (size_t)b * U_ * A_;
    const unsigned short* Kbase = Kb + (size_t)b * U_ * A_;

    v8f acc[8];
#pragma unroll
    for (int t = 0; t < 8; ++t) acc[t] = zero_v8f();

    for (int kc = 0; kc < A_; kc += 32) {
        const int k0 = kc + kbase;
        v16bf a = load_bf16_frag(Qbase + (size_t)(u0 + m) * A_ + k0);
#pragma unroll
        for (int t = 0; t < 8; ++t) {
            const int col0 = w * 128 + t * 16;
            v16bf bf = load_bf16_frag(Kbase + (size_t)(col0 + m) * A_ + k0);
            acc[t] = __builtin_amdgcn_wmma_f32_16x16x32_bf16(
                false, a, false, bf, (short)0, acc[t], false, false);
        }
    }

    // mask (reference multiplies masked logits by 1e-9, NOT -inf) and stage in LDS
    const int flags = nr_flags[b];
    const int ownn  = nr_own[b];
    const int enem  = nr_enemy[b];
#pragma unroll
    for (int t = 0; t < 8; ++t) {
        const int colg  = w * 128 + t * 16 + m;
        const bool colok = colg < enem;
#pragma unroll
        for (int r = 0; r < 8; ++r) {
            const int rowl = r + kbase;
            const int rowg = u0 + rowl;
            const bool ok = colok && (rowg >= flags) && (rowg < ownn);
            const float s = acc[t][r] * 0.0625f;    // 1/sqrt(256)
            sS[rowl * U_ + colg] = ok ? s : s * 1e-9f;
        }
    }
    __syncthreads();

    // cooperative row softmax: thread -> (row = tid/8, 64-col segment = tid%8)
    const int row = tid >> 3;
    const int seg = tid & 7;
    float* rp = sS + row * U_ + seg * 64;

    float mx = -3.0e38f;
#pragma unroll 8
    for (int c = 0; c < 64; ++c) mx = fmaxf(mx, rp[c]);
    sred[row * 8 + seg] = mx;
    __syncthreads();

    float rmax = sred[row * 8];
#pragma unroll
    for (int i = 1; i < 8; ++i) rmax = fmaxf(rmax, sred[row * 8 + i]);

    float sum = 0.0f;
#pragma unroll 8
    for (int c = 0; c < 64; ++c) {
        const float e = __expf(rp[c] - rmax);
        rp[c] = e;
        sum += e;
    }
    __syncthreads();                    // everyone done reading sred maxima
    sred[row * 8 + seg] = sum;
    __syncthreads();

    float tot = sred[row * 8];
#pragma unroll
    for (int i = 1; i < 8; ++i) tot += sred[row * 8 + i];
    const float inv = 1.0f / tot;

    float* op = out + ((size_t)(b * U_ + u0 + row)) * U_ + seg * 64;
#pragma unroll
    for (int c = 0; c < 64; c += 4) {
        f32x4 v;
        v[0] = rp[c]     * inv;
        v[1] = rp[c + 1] * inv;
        v[2] = rp[c + 2] * inv;
        v[3] = rp[c + 3] * inv;
        *(f32x4*)(op + c) = v;
    }
}

// ---------------- launch ----------------
extern "C" void kernel_launch(void* const* d_in, const int* in_sizes, int n_in,
                              void* d_out, int out_size, void* d_ws, size_t ws_size,
                              hipStream_t stream) {
    const float* are   = (const float*)d_in[0];
    const float* own   = (const float*)d_in[1];
    const float* enemy = (const float*)d_in[2];
    const int*   n_own = (const int*)d_in[3];
    const int*   n_en  = (const int*)d_in[4];
    const int*   n_fl  = (const int*)d_in[5];
    const float* Wq    = (const float*)d_in[6];
    const float* bq    = (const float*)d_in[7];
    const float* Wk    = (const float*)d_in[8];
    const float* bk    = (const float*)d_in[9];
    float* out = (float*)d_out;

    char* ws = (char*)d_ws;
    const size_t szWqT = (size_t)A_ * KQ_ * sizeof(unsigned short);     // 640 KiB
    const size_t szWkT = (size_t)A_ * UE_ * sizeof(unsigned short);     // 128 KiB
    const size_t szQb  = (size_t)B_ * U_ * A_ * sizeof(unsigned short); // 16 MiB
    unsigned short* WqT = (unsigned short*)(ws);
    unsigned short* WkT = (unsigned short*)(ws + szWqT);
    unsigned short* Qb  = (unsigned short*)(ws + szWqT + szWkT);
    unsigned short* Kb  = (unsigned short*)(ws + szWqT + szWkT + szQb);

    const int ntrans = A_ * KQ_ + A_ * UE_;   // 393216
    TargetHead_wtrans_kernel<<<(ntrans + 255) / 256, 256, 0, stream>>>(Wq, Wk, WqT, WkT);

    dim3 g1(B_ * (U_ / 16), 2);
    TargetHead_qk_gemm_kernel<<<g1, 32, 0, stream>>>(are, own, enemy, bq, bk,
                                                     WqT, WkT, Qb, Kb);

    TargetHead_attn_softmax_kernel<<<B_ * (U_ / 16), 128, 0, stream>>>(
        Qb, Kb, n_own, n_en, n_fl, out);
}
